// AttentionNet_20298015441442
// MI455X (gfx1250) — matrix operand order repaired and checked by reading, not compiled, not run on hardware
//
#include <hip/hip_runtime.h>
#include <hip/hip_bf16.h>

// ---------------------------------------------------------------------------
// AttentionNet on MI455X (gfx1250). All big GEMMs via v_wmma_f32_16x16x32_bf16.
// Fragment-friendly LDS layouts (all frag loads -> ds_load_b128, all staging ->
// 16B uint4 copies), pre-transposed weights, bf16 softmax tile, and TDM
// (tensor_load_to_lds, 6-arg form) staging of the K tile in the attn kernel.
// ---------------------------------------------------------------------------

typedef __attribute__((ext_vector_type(16))) __bf16 v16bf;
typedef __attribute__((ext_vector_type(8)))  float  v8f;
typedef __attribute__((ext_vector_type(4)))  unsigned int v4u;
typedef __attribute__((ext_vector_type(8)))  int v8i;
typedef __attribute__((ext_vector_type(4)))  int v4i;

#define B_   128
#define C_   2048
#define N_   196     // 14*14 spatial
#define NP_  224     // N padded to 7*32
#define S_   312
#define SP_  320
#define L_   300
#define M_   1024

#if __has_builtin(__builtin_amdgcn_tensor_load_to_lds) && \
    __has_builtin(__builtin_amdgcn_s_wait_tensorcnt)
#define USE_TDM 1
#else
#define USE_TDM 0
#endif

// ---- bf16 helpers ---------------------------------------------------------
static __device__ __forceinline__ unsigned short f2bf(float f) {
  unsigned int u = __float_as_uint(f);
  unsigned int r = (u + 0x7FFFu + ((u >> 16) & 1u)) >> 16;   // RNE
  return (unsigned short)r;
}
static __device__ __forceinline__ __bf16 bfbits(unsigned short u) {
  union { unsigned short s; __bf16 b; } c; c.s = u; return c.b;
}

// ---- WMMA fragment loaders (CDNA5 ISA 7.12.2, wave32) ---------------------
// A 16x32 bf16 (this code's k order, not the ISA table): lane<16 -> row=lane, K={0..15}; lane>=16 same row set,
// K={16..31}. One contiguous 32B run -> 2x ds_load_b128.
static __device__ __forceinline__ v16bf load_a_frag(const unsigned short* p,
                                                    int ldr, int lane) {
  int row = lane & 15;
  int kb  = (lane >> 4) << 4;
  const unsigned short* q = p + row * ldr + kb;
  v16bf a;
#pragma unroll
  for (int i = 0; i < 8; ++i) a[i] = bfbits(q[i]);
#pragma unroll
  for (int i = 0; i < 8; ++i) a[8 + i] = bfbits(q[8 + i]);
  return a;
}
// B 32x16 bf16, tile stored fragment-friendly as [col][k] (ldr in elements):
// lane -> col=lane&15, K run = (lane>>4)*16 .. +15: one 32B contiguous run.
static __device__ __forceinline__ v16bf load_b_frag_t(const unsigned short* p,
                                                      int ldr, int lane) {
  int col = lane & 15;
  int kb  = (lane >> 4) << 4;
  const unsigned short* q = p + col * ldr + kb;
  v16bf b;
#pragma unroll
  for (int i = 0; i < 16; ++i) b[i] = bfbits(q[i]);
  return b;
}
static __device__ __forceinline__ v8f wmma_bf16(v16bf a, v16bf b, v8f c) {
  return __builtin_amdgcn_wmma_f32_16x16x32_bf16(false, a, false, b,
                                                 (short)0, c, false, false);
}
// C/D 16x16 f32: col = lane%16, row = r + 8*(lane>>4).

// ---------------------------------------------------------------------------
// Kernel 1: feat [B,C,N] f32 -> featT [B,NP,C] bf16 (LDS tile transpose)
// ---------------------------------------------------------------------------
__global__ void __launch_bounds__(256) k_transpose(const float* __restrict__ feat,
                                                   unsigned short* __restrict__ featT) {
  __shared__ float tile[32][33];
  int b  = blockIdx.z;
  int c0 = blockIdx.y * 32;
  int n0 = blockIdx.x * 32;
  int tx = threadIdx.x, ty = threadIdx.y;      // 32 x 8
  const float* fb = feat + (size_t)b * C_ * N_;
#pragma unroll
  for (int i = 0; i < 4; ++i) {
    int n = n0 + tx;
    tile[ty + i * 8][tx] = (n < N_) ? fb[(size_t)(c0 + ty + i * 8) * N_ + n] : 0.0f;
  }
  __syncthreads();
  unsigned short* ob = featT + (size_t)b * NP_ * C_;
#pragma unroll
  for (int i = 0; i < 4; ++i)
    ob[(size_t)(n0 + ty + i * 8) * C_ + c0 + tx] = f2bf(tile[tx][ty + i * 8]);
}

// ---------------------------------------------------------------------------
// Kernel 2: weight transpose+convert: src [rows][cols] f32 -> dst [cols][rows] bf16
// ---------------------------------------------------------------------------
__global__ void __launch_bounds__(256) k_wtrans(const float* __restrict__ src,
                                                unsigned short* __restrict__ dst,
                                                int rows, int cols) {
  __shared__ float tile[32][33];
  int r0 = blockIdx.y * 32;
  int c0 = blockIdx.x * 32;
  int tx = threadIdx.x, ty = threadIdx.y;
#pragma unroll
  for (int i = 0; i < 4; ++i)
    tile[ty + i * 8][tx] = src[(size_t)(r0 + ty + i * 8) * cols + c0 + tx];
  __syncthreads();
#pragma unroll
  for (int i = 0; i < 4; ++i)
    dst[(size_t)(c0 + ty + i * 8) * rows + r0 + tx] = f2bf(tile[tx][ty + i * 8]);
}

// ---------------------------------------------------------------------------
// Kernel 3: feat_pool[b][c] = mean_n feat[b][c][n]
// ---------------------------------------------------------------------------
__global__ void __launch_bounds__(256) k_pool(const float* __restrict__ feat,
                                              float* __restrict__ pool) {
  int id = blockIdx.x * 256 + threadIdx.x;
  if (id >= B_ * C_) return;
  const float* p = feat + (size_t)id * N_;
  float s = 0.f;
  for (int n = 0; n < N_; ++n) s += p[n];
  pool[id] = s * (1.0f / N_);
}

// ---------------------------------------------------------------------------
// Kernel 4: Q = w2v_att @ Wq + bq  (0.2 GFLOP; plain fp32, bf16 out)
// ---------------------------------------------------------------------------
__global__ void __launch_bounds__(256) k_query(const float* __restrict__ w2v,
                                               const float* __restrict__ Wq,
                                               const float* __restrict__ bq,
                                               unsigned short* __restrict__ Q) {
  int id = blockIdx.x * 256 + threadIdx.x;
  if (id >= SP_ * M_) return;
  int s = id / M_, m = id - s * M_;
  float acc = 0.f;
  if (s < S_) {
    acc = bq[m];
    for (int l = 0; l < L_; ++l) acc += w2v[s * L_ + l] * Wq[l * M_ + m];
  }
  Q[id] = f2bf(acc);
}

// ---------------------------------------------------------------------------
// Kernel 5: key[b][n][m] (row-major) and valueT[b][m][n] (transposed) from
// featT [NP,C] x WkT/WvT [M,C] (+bias). Waves 0,1: key; waves 2,3: value.
// ---------------------------------------------------------------------------
__global__ void __launch_bounds__(256) k_kv(const unsigned short* __restrict__ featT,
                                            const unsigned short* __restrict__ WkT,
                                            const unsigned short* __restrict__ WvT,
                                            const float* __restrict__ bk,
                                            const float* __restrict__ bv,
                                            unsigned short* __restrict__ key,
                                            unsigned short* __restrict__ valueT) {
  __shared__ unsigned short As[32][32];     // A: [n_local][k]
  __shared__ unsigned short Bkt[64][32];    // B frag-friendly: [col][k]
  __shared__ unsigned short Bvt[64][32];
  int b  = blockIdx.z;
  int n0 = blockIdx.y * 32;
  int m0 = blockIdx.x * 64;
  int tid = threadIdx.x, lane = tid & 31, wave = tid >> 5;
  int trow = wave & 1;
  int isv  = wave >> 1;
  const unsigned short* fb = featT + (size_t)b * NP_ * C_;

  v8f acc[4] = {};
  for (int k0 = 0; k0 < C_; k0 += 32) {
    __builtin_prefetch(fb + (size_t)n0 * C_ + k0 + 128, 0, 1);
    if (tid < 128) {                        // A: 128 x 16B chunks
      int r = tid >> 2, k8 = (tid & 3) * 8;
      *(uint4*)&As[r][k8] = *(const uint4*)&fb[(size_t)(n0 + r) * C_ + k0 + k8];
    }
    {                                       // B tiles: 256 x 16B chunks each
      int cc = tid >> 2, k8 = (tid & 3) * 8;
      *(uint4*)&Bkt[cc][k8] = *(const uint4*)&WkT[(size_t)(m0 + cc) * C_ + k0 + k8];
      *(uint4*)&Bvt[cc][k8] = *(const uint4*)&WvT[(size_t)(m0 + cc) * C_ + k0 + k8];
    }
    __syncthreads();
    v16bf a = load_a_frag(&As[16 * trow][0], 32, lane);
    const unsigned short(*Bm)[32] = isv ? Bvt : Bkt;
#pragma unroll
    for (int t = 0; t < 4; ++t) {
      v16bf bb = load_b_frag_t(&Bm[16 * t][0], 32, lane);
      acc[t] = wmma_bf16(a, bb, acc[t]);
    }
    __syncthreads();
  }
  const float* bias = isv ? bv : bk;
#pragma unroll
  for (int t = 0; t < 4; ++t) {
    int col = m0 + 16 * t + (lane & 15);
    float bcol = bias[col];
#pragma unroll
    for (int r = 0; r < 8; ++r) {
      int n = n0 + 16 * trow + r + 8 * (lane >> 4);
      float v = (n < N_) ? (acc[t][r] + bcol) : 0.0f;   // zero pad rows
      if (!isv)
        key[((size_t)b * NP_ + n) * M_ + col] = f2bf(v);
      else
        valueT[((size_t)b * M_ + col) * NP_ + n] = f2bf(v);
    }
  }
}

// ---------------------------------------------------------------------------
// Kernel 6: scores = Q K^T -> softmax(n) -> attended = P V  (per b, 32 s-rows)
// K chunk staged via Tensor Data Mover when available.
// ---------------------------------------------------------------------------
__global__ void __launch_bounds__(256) k_attn(const unsigned short* __restrict__ Q,
                                              const unsigned short* __restrict__ key,
                                              const unsigned short* __restrict__ valueT,
                                              unsigned short* __restrict__ attended) {
  // LDS overlay: [Qs|Kt] (GEMM1) -> Sc (scores f32) -> Vt (GEMM2); P separate.
  __shared__ __align__(16) unsigned char smem[28672 + 14336];
  float          (*Sc)[NP_] = (float(*)[NP_])smem;                   // [32][224]
  unsigned short (*Qs)[32]  = (unsigned short(*)[32])smem;           // [32][32]
  unsigned short (*Kt)[32]  = (unsigned short(*)[32])(smem + 2048);  // [224][32]
  unsigned short (*Vt)[32]  = (unsigned short(*)[32])smem;           // [64][32]
  unsigned short (*P)[NP_]  = (unsigned short(*)[NP_])(smem + 28672);// [32][224]

  int s0 = blockIdx.x * 32;
  int b  = blockIdx.y;
  int tid = threadIdx.x, lane = tid & 31, wave = tid >> 5;
  int trow = wave & 1;
  int ngrp = wave >> 1;
  const unsigned short* kb = key    + (size_t)b * NP_ * M_;
  const unsigned short* vT = valueT + (size_t)b * M_ * NP_;

  // ---- GEMM1: scores[32 x 224], K = M = 1024 ----
  v8f acc[7] = {};
  for (int m0 = 0; m0 < M_; m0 += 32) {
    if (tid < 128) {                        // Qs: 128 x 16B chunks
      int r = tid >> 2, k8 = (tid & 3) * 8;
      *(uint4*)&Qs[r][k8] = *(const uint4*)&Q[(size_t)(s0 + r) * M_ + m0 + k8];
    }
#if USE_TDM
    // TDM: 2D tile (x=32 m-values, y=224 n-rows) from key[b][*][m0..m0+32)
    // lands row-major in LDS == Kt[n][ml]. Issued by wave 0 only (TDM ignores
    // EXEC, so gate at wave granularity). D# per CDNA5 ISA 8.3-8.4.
    if (wave == 0) {
      unsigned long long ga = (unsigned long long)(uintptr_t)(const void*)(kb + m0);
      unsigned int lds = (unsigned int)(uintptr_t)(void*)&Kt[0][0];
      v4u g0;
      g0[0] = 1u;                                            // count=1
      g0[1] = lds;                                           // lds_addr
      g0[2] = (unsigned int)(ga & 0xFFFFFFFFu);              // global_addr lo
      g0[3] = (unsigned int)((ga >> 32) & 0x01FFFFFFu) | 0x80000000u; // hi|type=2
      v8i g1;
      g1[0] = 0x00010000;                                    // data_size=2B
      g1[1] = (int)(32u << 16);                              // tensor_dim0=32
      g1[2] = (int)((unsigned)NP_ << 16);                    // tensor_dim1=224
      g1[3] = (int)(32u << 16);                              // tile_dim0=32
      g1[4] = NP_;                                           // tile_dim1=224
      g1[5] = M_;                                            // dim0_stride=1024
      g1[6] = 0; g1[7] = 0;
      v4i zz = {0, 0, 0, 0};
      v8i z8 = {0, 0, 0, 0, 0, 0, 0, 0};
      __builtin_amdgcn_tensor_load_to_lds(g0, g1, zz, zz, z8, 0);
      __builtin_amdgcn_s_wait_tensorcnt(0);
    }
#else
#pragma unroll
    for (int i = 0; i < 4; ++i) {           // Kt: 896 x 16B chunks
      int ch = tid + 256 * i;
      if (ch < 896) {
        int n = ch >> 2, m8 = (ch & 3) * 8;
        *(uint4*)&Kt[n][m8] = *(const uint4*)&kb[(size_t)n * M_ + m0 + m8];
      }
    }
#endif
    __syncthreads();
    v16bf a = load_a_frag(&Qs[16 * trow][0], 32, lane);
#pragma unroll
    for (int j = 0; j < 7; ++j) {
      int nt = 7 * ngrp + j;
      v16bf bb = load_b_frag_t(&Kt[16 * nt][0], 32, lane);
      acc[j] = wmma_bf16(a, bb, acc[j]);
    }
    __syncthreads();
  }
#pragma unroll
  for (int j = 0; j < 7; ++j) {            // spill scores (Qs/Kt dead)
    int col = 16 * (7 * ngrp + j) + (lane & 15);
#pragma unroll
    for (int r = 0; r < 8; ++r)
      Sc[16 * trow + r + 8 * (lane >> 4)][col] = acc[j][r];
  }
  __syncthreads();

  // ---- softmax over n, emit bf16 P (pad cols -> 0) ----
  if (tid < 32) {
    float mx = -3.0e38f;
    for (int n = 0; n < N_; ++n) mx = fmaxf(mx, Sc[tid][n]);
    float sum = 0.f;
    for (int n = 0; n < N_; ++n) {
      float e = __expf(Sc[tid][n] - mx);
      Sc[tid][n] = e;
      sum += e;
    }
    float inv = 1.0f / sum;
    for (int n = 0; n < N_; ++n) P[tid][n] = f2bf(Sc[tid][n] * inv);
    for (int n = N_; n < NP_; ++n) P[tid][n] = 0;
  }
  __syncthreads();

  // ---- GEMM2: attended[32 x M] = P[32 x 224] @ V[224 x M] ----
  int mt0 = 2 * (wave >> 1);
  for (int mc = 0; mc < M_; mc += 64) {
    v8f acc2[2] = {};
    for (int k0 = 0; k0 < NP_; k0 += 32) {
      {                                     // Vt: 256 x 16B chunks from valueT
        int cc = tid >> 2, k8 = (tid & 3) * 8;
        *(uint4*)&Vt[cc][k8] = *(const uint4*)&vT[(size_t)(mc + cc) * NP_ + k0 + k8];
      }
      __syncthreads();
      v16bf a = load_a_frag(&P[16 * trow][k0], NP_, lane);
#pragma unroll
      for (int t = 0; t < 2; ++t) {
        v16bf bb = load_b_frag_t(&Vt[16 * (mt0 + t)][0], 32, lane);
        acc2[t] = wmma_bf16(a, bb, acc2[t]);
      }
      __syncthreads();
    }
#pragma unroll
    for (int t = 0; t < 2; ++t) {
      int col = mc + 16 * (mt0 + t) + (lane & 15);
#pragma unroll
      for (int r = 0; r < 8; ++r) {
        int sl = 16 * trow + r + 8 * (lane >> 4);
        attended[((size_t)b * SP_ + s0 + sl) * M_ + col] = f2bf(acc2[t][r]);
      }
    }
  }
}

// ---------------------------------------------------------------------------
// Kernel 7: G = attended @ Wo ; v2s[b][s] = sum_c (pool+G+bo) * Vatt[s][c]
// ---------------------------------------------------------------------------
__global__ void __launch_bounds__(256) k_out(const unsigned short* __restrict__ attended,
                                             const unsigned short* __restrict__ WoT,
                                             const float* __restrict__ bo,
                                             const float* __restrict__ pool,
                                             const float* __restrict__ Vatt,
                                             float* __restrict__ out) {
  __shared__ unsigned short As[64][32];     // [s_local][k]
  __shared__ unsigned short Bt[64][32];     // [c_local][k]
  __shared__ float vpart[64];
  int s0 = blockIdx.x * 64;
  int b  = blockIdx.y;
  int tid = threadIdx.x, lane = tid & 31, wave = tid >> 5;
  if (tid < 64) vpart[tid] = 0.f;
  __syncthreads();

  float partial[8] = {0.f, 0.f, 0.f, 0.f, 0.f, 0.f, 0.f, 0.f};
  const unsigned short* ab = attended + ((size_t)b * SP_ + s0) * M_;

  for (int c0 = 0; c0 < C_; c0 += 64) {
    v8f acc[4] = {};
    for (int k0 = 0; k0 < M_; k0 += 32) {
      __builtin_prefetch(WoT + (size_t)c0 * M_ + k0 + 64, 0, 1);
      {                                     // As: 256 x 16B chunks
        int r = tid >> 2, k8 = (tid & 3) * 8;
        *(uint4*)&As[r][k8] = *(const uint4*)&ab[(size_t)r * M_ + k0 + k8];
      }
      {                                     // Bt: 256 x 16B chunks
        int cc = tid >> 2, k8 = (tid & 3) * 8;
        *(uint4*)&Bt[cc][k8] = *(const uint4*)&WoT[(size_t)(c0 + cc) * M_ + k0 + k8];
      }
      __syncthreads();
      v16bf a = load_a_frag(&As[16 * wave][0], 32, lane);
#pragma unroll
      for (int t = 0; t < 4; ++t) {
        v16bf bb = load_b_frag_t(&Bt[16 * t][0], 32, lane);
        acc[t] = wmma_bf16(a, bb, acc[t]);
      }
      __syncthreads();
    }
#pragma unroll
    for (int t = 0; t < 4; ++t) {           // fused (G+bo+pool)*Vatt epilogue
      int c = c0 + 16 * t + (lane & 15);
      float bc = bo[c];
      float pl = pool[(size_t)b * C_ + c];
#pragma unroll
      for (int r = 0; r < 8; ++r) {
        int s = s0 + 16 * wave + r + 8 * (lane >> 4);
        float g  = acc[t][r] + bc + pl;
        float va = (s < S_) ? Vatt[(size_t)s * C_ + c] : 0.f;
        partial[r] += g * va;
      }
    }
  }
#pragma unroll
  for (int r = 0; r < 8; ++r)
    atomicAdd(&vpart[16 * wave + r + 8 * (lane >> 4)], partial[r]);
  __syncthreads();
  if (tid < 64) {
    int s = s0 + tid;
    if (s < S_) out[(size_t)b * S_ + s] = vpart[tid];
  }
}

// ---------------------------------------------------------------------------
extern "C" void kernel_launch(void* const* d_in, const int* in_sizes, int n_in,
                              void* d_out, int out_size, void* d_ws, size_t ws_size,
                              hipStream_t stream) {
  (void)in_sizes; (void)n_in; (void)out_size; (void)ws_size;
  const float* feat = (const float*)d_in[0];
  const float* w2v  = (const float*)d_in[1];
  const float* Wq   = (const float*)d_in[2];
  const float* bq   = (const float*)d_in[3];
  const float* Wk   = (const float*)d_in[4];
  const float* bk   = (const float*)d_in[5];
  const float* Wv   = (const float*)d_in[6];
  const float* bv   = (const float*)d_in[7];
  const float* Wo   = (const float*)d_in[8];
  const float* bo   = (const float*)d_in[9];
  const float* Vatt = (const float*)d_in[10];
  float* out = (float*)d_out;

  char* ws = (char*)d_ws;
  size_t off = 0;
  auto alloc = [&](size_t bytes) -> char* {
    char* p = ws + off;
    off = (off + bytes + 255) & ~(size_t)255;
    return p;
  };
  unsigned short* featT  = (unsigned short*)alloc((size_t)B_ * NP_ * C_ * 2);
  unsigned short* keyb   = (unsigned short*)alloc((size_t)B_ * NP_ * M_ * 2);
  unsigned short* valueT = (unsigned short*)alloc((size_t)B_ * M_ * NP_ * 2);
  unsigned short* attnd  = (unsigned short*)alloc((size_t)B_ * SP_ * M_ * 2);
  unsigned short* Qb     = (unsigned short*)alloc((size_t)SP_ * M_ * 2);
  unsigned short* WkT    = (unsigned short*)alloc((size_t)M_ * C_ * 2);
  unsigned short* WvT    = (unsigned short*)alloc((size_t)M_ * C_ * 2);
  unsigned short* WoT    = (unsigned short*)alloc((size_t)C_ * M_ * 2);
  float*          poolb  = (float*)alloc((size_t)B_ * C_ * 4);

  k_transpose<<<dim3(NP_ / 32, C_ / 32, B_), dim3(32, 8), 0, stream>>>(feat, featT);
  k_pool<<<dim3((B_ * C_ + 255) / 256), 256, 0, stream>>>(feat, poolb);
  // Wk,Wv: [C][M] f32 -> [M][C] bf16 ; Wo: [M][C] f32 -> [C][M] bf16
  k_wtrans<<<dim3(M_ / 32, C_ / 32), dim3(32, 8), 0, stream>>>(Wk, WkT, C_, M_);
  k_wtrans<<<dim3(M_ / 32, C_ / 32), dim3(32, 8), 0, stream>>>(Wv, WvT, C_, M_);
  k_wtrans<<<dim3(C_ / 32, M_ / 32), dim3(32, 8), 0, stream>>>(Wo, WoT, M_, C_);
  k_query<<<dim3(SP_ * M_ / 256), 256, 0, stream>>>(w2v, Wq, bq, Qb);
  k_kv<<<dim3(M_ / 64, NP_ / 32, B_), 256, 0, stream>>>(featT, WkT, WvT, bk, bv,
                                                        keyb, valueT);
  k_attn<<<dim3(SP_ / 32, B_), 256, 0, stream>>>(Qb, keyb, valueT, attnd);
  k_out<<<dim3(SP_ / 64, B_), 256, 0, stream>>>(attnd, WoT, bo, poolb, Vatt, out);
}
